// ATSSAssigner_10127532884126
// MI455X (gfx1250) — compile-verified, matrix operand order checked by reading, not run on hardware
//
#include <hip/hip_runtime.h>
#include <hip/hip_bf16.h>
#include <math.h>

// ATSS assigner for MI455X (gfx1250, wave32).
//   per batch b:
//     kdist_wmma : d^2(gt, anchor) via v_wmma_f32_16x16x32_f16 (hi/lo f16 split -> f32-exact
//                  products), anchor-major dist buffer so each lane stores 2x b128.
//     ktopk      : per-level top-9 (lex (d,idx) == lax.top_k stability), thr = mean+std(ddof=1),
//                  center-in-gt test, positives -> atomic OR into u64 gt-bitmask per anchor.
//   kfinal       : per (b,anchor) resolve (popcount/ctz/argmax-IoU), gather targets, then
//                  LDS-staged fully-coalesced non-temporal sweeps for the one-hot score tensors.

#define ATSS_TOPK 9
#define ATSS_NPRO 31
#define ATSS_NALP 24
#define ATSS_NADS 37
#define NMAXC 64

typedef _Float16 v16h __attribute__((ext_vector_type(16)));
typedef float    v8f  __attribute__((ext_vector_type(8)));

// ---------------------------------------------------------------------------
// Kernel 1: squared center distances via WMMA, anchor-major output.
// d2[a][m] = |g_m|^2 + |a_a|^2 - 2 * (g_m . a_a)
// Cross term: A(16x32 f16) x B(32x16 f16), K0..7 = hi/lo split of (x,y):
//   A row m : [ghx, ghy, glx, gly, ghx, ghy, glx, gly]
//   B col n : [ahx, ahy, ahx, ahy, alx, aly, alx, aly]
//   sum_k   = (ghx+glx)(ahx+alx) + (ghy+gly)(ahy+aly) = gx*ax + gy*ay  (exact f16 products
//             accumulated in f32). One wave / 16-anchor tile; 4 unrolled WMMAs cover 64 rows.
// C-layout: lane holds 8 consecutive rows -> 2x float4 store per tile into dist[a*64 + m].
// ---------------------------------------------------------------------------
__global__ void __launch_bounds__(32)
kdist_wmma(const float* __restrict__ anc, const float* __restrict__ gtb,
           float* __restrict__ dist, int NA, int NMAX, int b) {
  __shared__ float s_gcx[NMAXC], s_gcy[NMAXC], s_gn2[NMAXC];
  const int lane = threadIdx.x;
  for (int m = lane; m < NMAXC; m += 32) {
    float cx = 0.f, cy = 0.f;
    if (m < NMAX) {
      const float* g = gtb + ((size_t)(b * NMAX + m)) * 4;
      cx = (g[0] + g[2]) * 0.5f; cy = (g[1] + g[3]) * 0.5f;
    }
    s_gcx[m] = cx; s_gcy[m] = cy; s_gn2[m] = cx * cx + cy * cy;
  }
  __syncthreads();

  const bool lo   = lane < 16;
  const int  c    = blockIdx.x * 16 + (lane & 15);
  const bool cval = c < NA;
  const int  cc   = cval ? c : NA - 1;
  const float* ab = anc + (size_t)cc * 4;
  float acx = (ab[0] + ab[2]) * 0.5f, acy = (ab[1] + ab[3]) * 0.5f;
  float an2 = acx * acx + acy * acy;

  _Float16 ahx = (_Float16)acx; _Float16 alx = (_Float16)(acx - (float)ahx);
  _Float16 ahy = (_Float16)acy; _Float16 aly = (_Float16)(acy - (float)ahy);
  v16h B;
#pragma unroll
  for (int e = 0; e < 16; ++e) B[e] = (_Float16)0.f;
  if (lo) { B[0]=ahx; B[1]=ahy; B[2]=ahx; B[3]=ahy; B[4]=alx; B[5]=aly; B[6]=alx; B[7]=aly; }

  float* dcol = dist + (size_t)c * NMAXC;  // only dereferenced when cval

#pragma unroll
  for (int mt = 0; mt < NMAXC / 16; ++mt) {
    const int m = mt * 16 + (lane & 15);
    float gx = s_gcx[m], gy = s_gcy[m];
    _Float16 ghx = (_Float16)gx; _Float16 glx = (_Float16)(gx - (float)ghx);
    _Float16 ghy = (_Float16)gy; _Float16 gly = (_Float16)(gy - (float)ghy);
    v16h A;
#pragma unroll
    for (int e = 0; e < 16; ++e) A[e] = (_Float16)0.f;
    if (lo) { A[0]=ghx; A[1]=ghy; A[2]=glx; A[3]=gly; A[4]=ghx; A[5]=ghy; A[6]=glx; A[7]=gly; }

    v8f C;
#pragma unroll
    for (int e = 0; e < 8; ++e) C[e] = 0.f;
    C = __builtin_amdgcn_wmma_f32_16x16x32_f16(false, A, false, B, (short)0, C, false, false);

    const int rbase = mt * 16 + (lo ? 0 : 8);
    float4 p0, p1;
    p0.x = s_gn2[rbase + 0] + an2 - 2.f * C[0];
    p0.y = s_gn2[rbase + 1] + an2 - 2.f * C[1];
    p0.z = s_gn2[rbase + 2] + an2 - 2.f * C[2];
    p0.w = s_gn2[rbase + 3] + an2 - 2.f * C[3];
    p1.x = s_gn2[rbase + 4] + an2 - 2.f * C[4];
    p1.y = s_gn2[rbase + 5] + an2 - 2.f * C[5];
    p1.z = s_gn2[rbase + 6] + an2 - 2.f * C[6];
    p1.w = s_gn2[rbase + 7] + an2 - 2.f * C[7];
    if (cval) {
      float4* p = (float4*)(dcol + rbase);
      p[0] = p0;
      p[1] = p1;
    }
  }
}

// ---------------------------------------------------------------------------
// Kernel 2: per (b, gt): per-level top-9 by (d, idx) lex order, candidate IoU,
// thr = mean + std(ddof=1), positives scattered into bits[b*NA + anchor].
// dist is anchor-major (stride NMAXC); buffer is L2-resident (8.6 MB << 192 MB).
// ---------------------------------------------------------------------------
__global__ void __launch_bounds__(32)
ktopk(const float* __restrict__ dist, const float* __restrict__ anc,
      const float* __restrict__ gtb, const float* __restrict__ mgt,
      const int* __restrict__ nlev, unsigned long long* __restrict__ bits,
      int NA, int NLVL, int NMAX, int b) {
  const int lane = threadIdx.x;
  const int g = blockIdx.x;

  __shared__ float sd[32 * ATSS_TOPK];
  __shared__ int   si[32 * ATSS_TOPK];
  __shared__ int   cand[32];
  __shared__ float s_red[32], s_red2[32];
  __shared__ float s_thr;

  int nc = 0, start = 0;
  for (int lv = 0; lv < NLVL; ++lv) {
    const int L = nlev[lv];
    const int k = L < ATSS_TOPK ? L : ATSS_TOPK;
    float bd[ATSS_TOPK]; int bi[ATSS_TOPK];
#pragma unroll
    for (int j = 0; j < ATSS_TOPK; ++j) { bd[j] = 3.4e38f; bi[j] = 0x7fffffff; }
    for (int i = lane; i < L; i += 32) {
      const int ix = start + i;
      const float dv = dist[(size_t)ix * NMAXC + g];
      if (dv < bd[ATSS_TOPK - 1] ||
          (dv == bd[ATSS_TOPK - 1] && ix < bi[ATSS_TOPK - 1])) {
        bd[ATSS_TOPK - 1] = dv; bi[ATSS_TOPK - 1] = ix;
#pragma unroll
        for (int j = ATSS_TOPK - 1; j > 0; --j) {
          bool sw = (bd[j] < bd[j - 1]) || (bd[j] == bd[j - 1] && bi[j] < bi[j - 1]);
          if (sw) {
            float tf = bd[j]; bd[j] = bd[j - 1]; bd[j - 1] = tf;
            int   ti = bi[j]; bi[j] = bi[j - 1]; bi[j - 1] = ti;
          }
        }
      }
    }
#pragma unroll
    for (int j = 0; j < ATSS_TOPK; ++j) { sd[lane * ATSS_TOPK + j] = bd[j]; si[lane * ATSS_TOPK + j] = bi[j]; }
    __syncthreads();
    if (lane == 0) {
      for (int s = 0; s < k; ++s) {
        float bv = 3.4e38f; int bx = 0x7fffffff; int bj = -1;
        for (int j = 0; j < 32 * ATSS_TOPK; ++j) {
          float dv = sd[j]; int ix = si[j];
          if (dv < bv || (dv == bv && ix < bx)) { bv = dv; bx = ix; bj = j; }
        }
        cand[nc + s] = bx;
        sd[bj] = 3.4e38f; si[bj] = 0x7fffffff;
      }
    }
    __syncthreads();
    nc += k; start += L;
  }

  // Candidate IoU (pairwise_iou: union = max(a1+a2-ov, 1e-6)) + center-in-gt test.
  const float* gb = gtb + (size_t)(b * NMAX + g) * 4;
  const float gx1 = gb[0], gy1 = gb[1], gx2 = gb[2], gy2 = gb[3];
  float ov = 0.f; bool ing = false; int ci = 0;
  if (lane < nc) {
    ci = cand[lane];
    const float* ab = anc + (size_t)ci * 4;
    float ax1 = ab[0], ay1 = ab[1], ax2 = ab[2], ay2 = ab[3];
    float a1 = (gx2 - gx1) * (gy2 - gy1);
    float a2 = (ax2 - ax1) * (ay2 - ay1);
    float w = fmaxf(fminf(gx2, ax2) - fmaxf(gx1, ax1), 0.f);
    float h = fmaxf(fminf(gy2, ay2) - fmaxf(gy1, ay1), 0.f);
    float inter = w * h;
    ov = inter / fmaxf(a1 + a2 - inter, 1e-6f);
    float acx = (ax1 + ax2) * 0.5f, acy = (ay1 + ay2) * 0.5f;
    float mn = fminf(fminf(acx - gx1, acy - gy1), fminf(gx2 - acx, gy2 - acy));
    ing = mn > 1e-9f;
  }
  s_red[lane]  = (lane < nc) ? ov : 0.f;
  s_red2[lane] = (lane < nc) ? ov * ov : 0.f;
  __syncthreads();
  if (lane == 0) {
    float s = 0.f, q = 0.f;
    for (int j = 0; j < 32; ++j) { s += s_red[j]; q += s_red2[j]; }
    float n = (float)nc;
    float mean = s / n;
    float var = (q - s * s / n) / fmaxf(n - 1.f, 1.f);
    s_thr = mean + sqrtf(fmaxf(var, 0.f));
  }
  __syncthreads();
  const float mg = mgt[b * NMAX + g];
  if (lane < nc && ing && mg > 0.f && ov > s_thr)
    atomicOr(&bits[(size_t)b * NA + ci], 1ull << g);
}

// ---------------------------------------------------------------------------
// Kernel 3: per (b, anchor) resolve + gather; LDS-staged coalesced score writes.
// Output pass is the roofline-critical 320 MB -> non-temporal, coalesced stores.
// ---------------------------------------------------------------------------
__global__ void __launch_bounds__(256)
kfinal(const float* __restrict__ anc, const float* __restrict__ gpro,
       const float* __restrict__ galp, const float* __restrict__ gads,
       const float* __restrict__ gtb, const float* __restrict__ gcor,
       const float* __restrict__ pd, const unsigned long long* __restrict__ bits,
       float* __restrict__ out, int NA, int NMAX) {
  const int b   = blockIdx.y;
  const int BS  = gridDim.y;
  const int a0  = blockIdx.x * 256;
  const int tid = threadIdx.x;
  const int a   = a0 + tid;
  const int nv  = (NA - a0) < 256 ? (NA - a0) : 256;

  __shared__ float sgt[NMAXC * 4], scor[NMAXC * 8], spro[NMAXC], salp[NMAXC], sads[NMAXC * 6];
  __shared__ int   s_tp[256], s_ta[256], s_tad[6 * 256];
  __shared__ float s_iou[256];

  for (int t = tid; t < NMAX * 4; t += 256) sgt[t]  = gtb[(size_t)b * NMAX * 4 + t];
  for (int t = tid; t < NMAX * 8; t += 256) scor[t] = gcor[(size_t)b * NMAX * 8 + t];
  for (int t = tid; t < NMAX * 6; t += 256) sads[t] = gads[(size_t)b * NMAX * 6 + t];
  for (int t = tid; t < NMAX;     t += 256) { spro[t] = gpro[b * NMAX + t]; salp[t] = galp[b * NMAX + t]; }
  __syncthreads();

  const size_t BA = (size_t)BS * NA;
  const size_t i  = (size_t)b * NA + a;

  if (a < NA) {
    unsigned long long m = bits[i];
    const int fg = __popcll(m);
    const bool fgb = fg > 0;
    int tg = 0;
    if (fg == 1) {
      tg = (int)__builtin_ctzll(m);               // argmax of one-hot == lowest set bit
    } else if (fg > 1) {
      // multi-assigned: gt with max pairwise IoU (first max wins, like argmax)
      const float* ab = anc + (size_t)a * 4;
      float ax1 = ab[0], ay1 = ab[1], ax2 = ab[2], ay2 = ab[3];
      float aa = (ax2 - ax1) * (ay2 - ay1);
      float best = -1.f; int bg = 0;
      for (int gi = 0; gi < NMAX; ++gi) {
        float gx1 = sgt[gi * 4], gy1 = sgt[gi * 4 + 1], gx2 = sgt[gi * 4 + 2], gy2 = sgt[gi * 4 + 3];
        float a1 = (gx2 - gx1) * (gy2 - gy1);
        float w = fmaxf(fminf(gx2, ax2) - fmaxf(gx1, ax1), 0.f);
        float h = fmaxf(fminf(gy2, ay2) - fmaxf(gy1, ay1), 0.f);
        float inter = w * h;
        float ovv = inter / fmaxf(a1 + aa - inter, 1e-6f);
        if (ovv > best) { best = ovv; bg = gi; }
      }
      tg = bg;
    }

    // ious = max_g( batched_iou(gt, pd) * mask_pos ) == IoU(selected gt, pred box)
    float sel_iou = 0.f;
    if (fgb) {
      const float* pb = pd + i * 4;
      float px1 = pb[0], py1 = pb[1], px2 = pb[2], py2 = pb[3];
      float gx1 = sgt[tg * 4], gy1 = sgt[tg * 4 + 1], gx2 = sgt[tg * 4 + 2], gy2 = sgt[tg * 4 + 3];
      float w = fmaxf(fminf(gx2, px2) - fmaxf(gx1, px1), 0.f);
      float h = fmaxf(fminf(gy2, py2) - fmaxf(gy1, py1), 0.f);
      float inter = w * h;
      float a1 = fmaxf(gx2 - gx1, 0.f) * fmaxf(gy2 - gy1, 0.f);
      float a2 = fmaxf(px2 - px1, 0.f) * fmaxf(py2 - py1, 0.f);
      sel_iou = inter / (a1 + a2 - inter + 1e-9f);
    }

    // Sentinel class (== num_classes) never matches a column -> background rows write 0.
    const int tp = fgb ? (int)spro[tg] : ATSS_NPRO;
    const int ta = fgb ? (int)salp[tg] : ATSS_NALP;
    s_tp[tid] = tp; s_ta[tid] = ta; s_iou[tid] = sel_iou;
#pragma unroll
    for (int j = 0; j < 6; ++j) {
      int tv = fgb ? (int)sads[tg * 6 + j] : ATSS_NADS;
      s_tad[j * 256 + tid] = tv;
      __builtin_nontemporal_store((float)tv, &out[(2 + j) * BA + i]);   // target_ad0..5
    }
    __builtin_nontemporal_store((float)tp, &out[i]);                    // target_pro
    __builtin_nontemporal_store((float)ta, &out[BA + i]);               // target_alp

    float* ob = out + 8 * BA + i * 4;                                   // target_bboxes (contig/thread)
#pragma unroll
    for (int j = 0; j < 4; ++j) __builtin_nontemporal_store(sgt[tg * 4 + j], &ob[j]);
    float* oc = out + 12 * BA + i * 8;                                  // target_corners
#pragma unroll
    for (int j = 0; j < 8; ++j) __builtin_nontemporal_store(scor[tg * 8 + j], &oc[j]);

    __builtin_nontemporal_store(fgb ? 1.f : 0.f, &out[297 * BA + i]);   // fg mask (bool)
  }
  __syncthreads();

  // Coalesced one-hot score sweeps: consecutive lanes -> consecutive addresses.
  const size_t i0 = (size_t)b * NA + a0;
  {
    float* base = out + 20 * BA + i0 * ATSS_NPRO;                       // pro_scores
    const int total = nv * ATSS_NPRO;
    for (int t = tid; t < total; t += 256) {
      int j = t / ATSS_NPRO, c = t - j * ATSS_NPRO;
      __builtin_nontemporal_store((c == s_tp[j]) ? s_iou[j] : 0.f, &base[t]);
    }
  }
  {
    float* base = out + 51 * BA + i0 * ATSS_NALP;                       // alp_scores
    const int total = nv * ATSS_NALP;
    for (int t = tid; t < total; t += 256) {
      int j = t / ATSS_NALP, c = t - j * ATSS_NALP;
      __builtin_nontemporal_store((c == s_ta[j]) ? s_iou[j] : 0.f, &base[t]);
    }
  }
#pragma unroll
  for (int jj = 0; jj < 6; ++jj) {                                      // ads_scores 0..5
    float* base = out + (size_t)(75 + 37 * jj) * BA + i0 * ATSS_NADS;
    const int total = nv * ATSS_NADS;
    for (int t = tid; t < total; t += 256) {
      int j = t / ATSS_NADS, c = t - j * ATSS_NADS;
      __builtin_nontemporal_store((c == s_tad[jj * 256 + j]) ? s_iou[j] : 0.f, &base[t]);
    }
  }
}

// ---------------------------------------------------------------------------
extern "C" void kernel_launch(void* const* d_in, const int* in_sizes, int n_in,
                              void* d_out, int out_size, void* d_ws, size_t ws_size,
                              hipStream_t stream) {
  const float* anc  = (const float*)d_in[0];
  const int*   nlev = (const int*)  d_in[1];
  const float* gpro = (const float*)d_in[2];
  const float* galp = (const float*)d_in[3];
  const float* gads = (const float*)d_in[4];
  const float* gtb  = (const float*)d_in[5];
  const float* gcor = (const float*)d_in[6];
  const float* mgt  = (const float*)d_in[7];
  const float* pd   = (const float*)d_in[8];

  const int NA   = in_sizes[0] / 4;            // 33600
  const int NLVL = in_sizes[1];                // 3
  const int BS   = in_sizes[8] / (NA * 4);     // 8
  int NMAX = in_sizes[5] / (BS * 4);           // 64
  if (NMAX > NMAXC) NMAX = NMAXC;              // u64 gt-bitmask bound

  // Workspace: [dist (NA x 64 f32, anchor-major, reused per batch)] [bits (BS x NA u64)]
  float* dist = (float*)d_ws;
  const size_t distBytes = (((size_t)NA * NMAXC * sizeof(float)) + 255) & ~(size_t)255;
  unsigned long long* bits = (unsigned long long*)((char*)d_ws + distBytes);
  hipMemsetAsync(bits, 0, (size_t)BS * NA * sizeof(unsigned long long), stream);

  for (int b = 0; b < BS; ++b) {
    kdist_wmma<<<dim3((NA + 15) / 16), dim3(32), 0, stream>>>(anc, gtb, dist, NA, NMAX, b);
    ktopk<<<dim3(NMAX), dim3(32), 0, stream>>>(dist, anc, gtb, mgt, nlev, bits, NA, NLVL, NMAX, b);
  }
  kfinal<<<dim3((NA + 255) / 256, BS), dim3(256), 0, stream>>>(
      anc, gpro, galp, gads, gtb, gcor, pd, bits, (float*)d_out, NA, NMAX);
}
